// global_pool_att_81475529605236
// MI455X (gfx1250) — compile-verified, matrix operand order checked by reading, not run on hardware
//
#include <hip/hip_runtime.h>
#include <hip/hip_bf16.h>

typedef __attribute__((ext_vector_type(16))) __bf16 v16bf;
typedef __attribute__((ext_vector_type(8)))  float  v8f;

#define DIMV 128
#define EPSV 1e-5f

// Ordered-uint encoding of float so unsigned atomicMax == float max.
__device__ __forceinline__ unsigned ordf(float f) {
    unsigned u = __float_as_uint(f);
    return (u & 0x80000000u) ? ~u : (u | 0x80000000u);
}
__device__ __forceinline__ float unordf(unsigned u) {
    return (u & 0x80000000u) ? __uint_as_float(u & 0x7FFFFFFFu) : __uint_as_float(~u);
}

// ---------------- kernel 0: zero init ----------------
__global__ void k_init(float* __restrict__ out, float* __restrict__ denom,
                       unsigned* __restrict__ mbits, int B) {
    int i = blockIdx.x * blockDim.x + threadIdx.x;
    if (i < B * DIMV) out[i] = 0.f;
    if (i < B) { denom[i] = 0.f; mbits[i] = 0u; }
}

// ---------------- kernel 1: q = LN(x[keypoints]) in bf16 ----------------
__global__ void __launch_bounds__(256)
k_query(const float* __restrict__ x, const float* __restrict__ gamma,
        const float* __restrict__ beta, const int* __restrict__ keypoints,
        __bf16* __restrict__ q, int B) {
    int wave = threadIdx.x >> 5, lane = threadIdx.x & 31;
    int b = blockIdx.x * 8 + wave;
    if (b >= B) return;
    const float* row = x + (size_t)keypoints[b] * DIMV;
    float4 v = *(const float4*)(row + lane * 4);
    float s  = v.x + v.y + v.z + v.w;
    float ss = v.x*v.x + v.y*v.y + v.z*v.z + v.w*v.w;
    #pragma unroll
    for (int off = 16; off >= 1; off >>= 1) { s += __shfl_xor(s, off); ss += __shfl_xor(ss, off); }
    float mu = s * (1.f / DIMV);
    float rs = rsqrtf(ss * (1.f / DIMV) - mu * mu + EPSV);
    float4 g  = *(const float4*)(gamma + lane * 4);
    float4 bt = *(const float4*)(beta  + lane * 4);
    __bf16* qr = q + (size_t)b * DIMV + lane * 4;
    qr[0] = (__bf16)((v.x - mu) * rs * g.x + bt.x);
    qr[1] = (__bf16)((v.y - mu) * rs * g.y + bt.y);
    qr[2] = (__bf16)((v.z - mu) * rs * g.z + bt.z);
    qr[3] = (__bf16)((v.w - mu) * rs * g.w + bt.w);
}

// ---------------- kernel 2: scores via WMMA diagonal, segment max ----------------
// Per wave: 16-node tile. A frag = LN'd rows in bf16 A-layout; B frag column j =
// q[batch[node_j]]; scores are the diagonal of the 16x16 f32 accumulator.
__global__ void __launch_bounds__(256)
k_scores(const float* __restrict__ x, const float* __restrict__ gamma,
         const float* __restrict__ beta, const int* __restrict__ batch,
         const __bf16* __restrict__ q, float* __restrict__ scores,
         unsigned* __restrict__ mbits, int N) {
    __shared__ float g_s[DIMV], bt_s[DIMV];
    if (threadIdx.x < DIMV) { g_s[threadIdx.x] = gamma[threadIdx.x]; bt_s[threadIdx.x] = beta[threadIdx.x]; }
    __syncthreads();

    int wave = threadIdx.x >> 5, lane = threadIdx.x & 31;
    int t0 = (blockIdx.x * 8 + wave) * 16;              // 16 nodes / wave, 128 / block
    if (t0 + 16 > N) return;
    int half = lane >> 4;                               // which half of K blocks this lane holds
    int node = t0 + (lane & 15);
    const float* xrow = x + (size_t)node * DIMV;
    if (node + 128 < N) __builtin_prefetch(xrow + 128 * DIMV, 0, 0);  // global_prefetch_b8

    // Load the 64 K-values this lane owns: per 32-chunk c, blocks [b1,b1+7],[b1+16,b1+23].
    float vals[64];
    float s = 0.f, ss = 0.f;
    #pragma unroll
    for (int c = 0; c < 4; ++c) {
        int b1 = c * 32 + half * 8;
        float4 p0 = *(const float4*)(xrow + b1);
        float4 p1 = *(const float4*)(xrow + b1 + 4);
        float4 p2 = *(const float4*)(xrow + b1 + 16);
        float4 p3 = *(const float4*)(xrow + b1 + 20);
        float* vp = &vals[c * 16];
        vp[0]=p0.x; vp[1]=p0.y; vp[2]=p0.z;  vp[3]=p0.w;
        vp[4]=p1.x; vp[5]=p1.y; vp[6]=p1.z;  vp[7]=p1.w;
        vp[8]=p2.x; vp[9]=p2.y; vp[10]=p2.z; vp[11]=p2.w;
        vp[12]=p3.x; vp[13]=p3.y; vp[14]=p3.z; vp[15]=p3.w;
        #pragma unroll
        for (int j = 0; j < 16; ++j) { s += vp[j]; ss += vp[j] * vp[j]; }
    }
    // lanes L and L+16 together hold the full row
    s += __shfl_xor(s, 16); ss += __shfl_xor(ss, 16);
    float mu = s * (1.f / DIMV);
    float rs = rsqrtf(ss * (1.f / DIMV) - mu * mu + EPSV);

    int seg = batch[node];
    const __bf16* qrow = q + (size_t)seg * DIMV;

    v8f acc = {};
    #pragma unroll
    for (int c = 0; c < 4; ++c) {
        int b1 = c * 32 + half * 8;
        v16bf A, Bf;
        #pragma unroll
        for (int j = 0; j < 8; ++j) {
            int k1 = b1 + j, k2 = b1 + 16 + j;
            A[j]     = (__bf16)((vals[c * 16 + j]     - mu) * rs * g_s[k1] + bt_s[k1]);
            A[j + 8] = (__bf16)((vals[c * 16 + 8 + j] - mu) * rs * g_s[k2] + bt_s[k2]);
            Bf[j]     = qrow[k1];
            Bf[j + 8] = qrow[k2];
        }
        acc = __builtin_amdgcn_wmma_f32_16x16x32_bf16(false, A, false, Bf,
                                                      (short)0, acc, false, false);
    }

    // Diagonal of D: (j,j) for j<8 lives in lane j / vgpr j; for j>=8 in lane j+16 / vgpr j-8.
    float sc = 0.f; int myNode = -1;
    if (lane < 8)        { sc = acc[lane];      myNode = t0 + lane; }
    else if (lane >= 24) { sc = acc[lane - 24]; myNode = t0 + (lane - 16); }
    if (myNode >= 0) {
        scores[myNode] = sc;
        atomicMax(&mbits[batch[myNode]], ordf(sc));
    }
}

// ---------------- kernel 3: e = exp(s - m), denom = segment_sum(e) ----------------
__global__ void __launch_bounds__(256)
k_expsum(const int* __restrict__ batch, const unsigned* __restrict__ mbits,
         float* __restrict__ scores, float* __restrict__ denom, int N) {
    int i = blockIdx.x * blockDim.x + threadIdx.x;
    if (i >= N) return;
    int b = batch[i];
    float e = __expf(scores[i] - unordf(mbits[b]));
    scores[i] = e;                                   // scores buffer now holds e
    int b0 = __shfl(b, 0);
    if (__all(b == b0)) {                            // sorted segments: common case
        float t = e;
        #pragma unroll
        for (int off = 16; off >= 1; off >>= 1) t += __shfl_xor(t, off);
        if ((threadIdx.x & 31) == 0) atomicAdd(&denom[b], t);
    } else {
        atomicAdd(&denom[b], e);
    }
}

// ---------------- kernel 4: out[b] = segment_sum(xn * w) ----------------
// x rows are staged into LDS with CDNA5 async global->LDS DMA (ASYNCcnt path),
// normalized in place, then 128 channel threads do run-length segmented pooling.
__global__ void __launch_bounds__(256)
k_pool(const float* __restrict__ x, const float* __restrict__ gamma,
       const float* __restrict__ beta, const int* __restrict__ batch,
       const float* __restrict__ e, const float* __restrict__ denom,
       float* __restrict__ out, int N) {
    constexpr int C = 64;                 // nodes per block
    __shared__ float xn_s[C][DIMV];       // 32 KB: raw rows, normalized in place
    __shared__ float w_s[C];
    __shared__ int   seg_s[C];
    __shared__ float g_s[DIMV], bt_s[DIMV];
    int n0 = blockIdx.x * C;
    if (n0 + C > N) return;
    if (threadIdx.x < DIMV) { g_s[threadIdx.x] = gamma[threadIdx.x]; bt_s[threadIdx.x] = beta[threadIdx.x]; }

    int wave = threadIdx.x >> 5, lane = threadIdx.x & 31;

    // Phase A0: async DMA this wave's 8 rows (512 B each) into LDS, 8 in flight.
    #pragma unroll
    for (int it = 0; it < 8; ++it) {
        int li = wave * 8 + it;
        const float* gsrc = x + (size_t)(n0 + li) * DIMV + lane * 4;   // 16 B per lane
        unsigned ldsdst = (unsigned)(uintptr_t)&xn_s[li][lane * 4];    // wave-relative LDS byte addr
        asm volatile("global_load_async_to_lds_b128 %0, %1, off"
                     :: "v"(ldsdst), "v"(gsrc) : "memory");
    }
    asm volatile("s_wait_asynccnt 0x0" ::: "memory");   // wave's rows resident in LDS
    __syncthreads();                                    // g_s/bt_s ready, all rows staged

    // Phase A1: LayerNorm each row in place in LDS.
    #pragma unroll
    for (int it = 0; it < 8; ++it) {
        int li = wave * 8 + it;
        int node = n0 + li;
        float4 v = *(float4*)&xn_s[li][lane * 4];
        float s  = v.x + v.y + v.z + v.w;
        float ss = v.x*v.x + v.y*v.y + v.z*v.z + v.w*v.w;
        #pragma unroll
        for (int off = 16; off >= 1; off >>= 1) { s += __shfl_xor(s, off); ss += __shfl_xor(ss, off); }
        float mu = s * (1.f / DIMV);
        float rs = rsqrtf(ss * (1.f / DIMV) - mu * mu + EPSV);
        int k = lane * 4;
        float4 xn;
        xn.x = (v.x - mu) * rs * g_s[k + 0] + bt_s[k + 0];
        xn.y = (v.y - mu) * rs * g_s[k + 1] + bt_s[k + 1];
        xn.z = (v.z - mu) * rs * g_s[k + 2] + bt_s[k + 2];
        xn.w = (v.w - mu) * rs * g_s[k + 3] + bt_s[k + 3];
        *(float4*)&xn_s[li][k] = xn;
        if (lane == 0) {
            int sg = batch[node];
            float d = denom[sg];
            w_s[li] = e[node] / (d > 0.f ? d : 1.f);
            seg_s[li] = sg;
        }
    }
    __syncthreads();

    // Phase B: 128 channel threads, run-length segmented accumulation.
    if (threadIdx.x < DIMV) {
        int k = threadIdx.x;
        float acc = 0.f;
        int cur = seg_s[0];
        for (int i = 0; i < C; ++i) {
            int sg = seg_s[i];
            if (sg != cur) {
                atomicAdd(&out[(size_t)cur * DIMV + k], acc);
                acc = 0.f; cur = sg;
            }
            acc += xn_s[i][k] * w_s[i];
        }
        atomicAdd(&out[(size_t)cur * DIMV + k], acc);
    }
}

// ---------------- host launch ----------------
extern "C" void kernel_launch(void* const* d_in, const int* in_sizes, int n_in,
                              void* d_out, int out_size, void* d_ws, size_t ws_size,
                              hipStream_t stream) {
    const float* x        = (const float*)d_in[0];
    const float* gamma    = (const float*)d_in[1];
    const float* beta     = (const float*)d_in[2];
    const int*   batch    = (const int*)d_in[3];
    const int*   keypts   = (const int*)d_in[4];
    const int N = in_sizes[3];
    const int B = in_sizes[4];
    float* out = (float*)d_out;

    // workspace carve-out (~5.1 MB): q bf16 | scores/e f32 | mbits u32 | denom f32
    char* ws = (char*)d_ws;
    __bf16* q = (__bf16*)ws;
    size_t off = ((size_t)B * DIMV * sizeof(__bf16) + 255) & ~(size_t)255;
    float* scores = (float*)(ws + off);
    off += ((size_t)N * sizeof(float) + 255) & ~(size_t)255;
    unsigned* mbits = (unsigned*)(ws + off);
    off += ((size_t)B * sizeof(unsigned) + 255) & ~(size_t)255;
    float* denom = (float*)(ws + off);

    k_init  <<<(B * DIMV + 255) / 256, 256, 0, stream>>>(out, denom, mbits, B);
    k_query <<<(B + 7) / 8,            256, 0, stream>>>(x, gamma, beta, keypts, q, B);
    k_scores<<<(N + 127) / 128,        256, 0, stream>>>(x, gamma, beta, batch, q, scores, mbits, N);
    k_expsum<<<(N + 255) / 256,        256, 0, stream>>>(batch, mbits, scores, denom, N);
    k_pool  <<<(N + 63) / 64,          256, 0, stream>>>(x, gamma, beta, batch, scores, denom, out, N);
}
